// EncoderBlock_38053410242971
// MI455X (gfx1250) — compile-verified, matrix operand order checked by reading, not run on hardware
//
#include <hip/hip_runtime.h>
#include <cstdint>
#include <cstddef>

// ---------------- problem dims ----------------
#define DM    1024          // d_model
#define NH    16            // heads
#define DKH   64            // d_k
#define DFF   4096
#define SEQ   2048
#define BATCH 2
#define ROWS  (BATCH*SEQ)   // 4096 tokens

typedef __bf16 bf16;
typedef __attribute__((ext_vector_type(16))) __bf16 v16bf;
typedef __attribute__((ext_vector_type(8)))  float  v8f;

union AFrag { v16bf v; bf16  e[16]; };
union CFrag { v8f   v; float e[8];  };

__device__ inline v8f zero_v8f() {
    v8f z = {0.f, 0.f, 0.f, 0.f, 0.f, 0.f, 0.f, 0.f};
    return z;
}

// ------------------------------------------------------------------
// CDNA5 async global->LDS copy (ASYNCcnt tracked, no VGPR staging).
// lds_off : per-lane LDS byte offset (flat LDS addr low 32 bits)
// voff    : per-lane 32-bit byte offset added to the 64-bit SGPR base
// ------------------------------------------------------------------
__device__ inline void async_copy_b128(uint32_t lds_off, uint32_t voff, uint64_t sbase) {
    asm volatile("global_load_async_to_lds_b128 %0, %1, %2"
                 :: "v"(lds_off), "v"(voff), "s"(sbase) : "memory");
}
__device__ inline void wait_async0() {
    asm volatile("s_wait_asynccnt 0x0" ::: "memory");
}
__device__ inline uint32_t lds_offset_of(const void* p) {
    return (uint32_t)(uintptr_t)p;   // LDS flat addr keeps offset in [31:0]
}

// ------------------------------------------------------------------
// Load a 16x32 bf16 fragment in the CDNA5 WMMA A/B operand layout.
// A (16Mx32K): lane L holds row m = L&15; VGPRs 0..3 hold K pairs
// {kb..kb+7}, VGPRs 4..7 hold {16+kb..16+kb+7} where kb = (L>=16)?8:0.
// The B operand (32Kx16N) has the identical pattern with N in place of
// M, so a [N][K] (K-contiguous) tile uses the same loader.
// Each half (e[0..7], e[8..15]) is 16 contiguous bytes -> 2x b128 loads.
// ------------------------------------------------------------------
__device__ inline v16bf load_frag16x32(const bf16* __restrict__ base, int ld) {
    const int lane = threadIdx.x & 31;
    const int m    = lane & 15;
    const int kb   = (lane & 16) ? 8 : 0;
    const bf16* row = base + (size_t)m * ld;
    AFrag a;
#pragma unroll
    for (int i = 0; i < 4; ++i) {
        a.e[2*i]     = row[kb + 2*i];
        a.e[2*i + 1] = row[kb + 2*i + 1];
        a.e[8 + 2*i]     = row[16 + kb + 2*i];
        a.e[8 + 2*i + 1] = row[16 + kb + 2*i + 1];
    }
    return a.v;
}

// ------------------------------------------------------------------
// fp32 [K][N] weight -> bf16 [N][K] (K-contiguous) for B operands.
// ------------------------------------------------------------------
__global__ __launch_bounds__(256)
void enc_transpose_bf16(const float* __restrict__ W, bf16* __restrict__ Wt,
                        int K, int N) {
    __shared__ float tile[32][33];
    const int n0 = blockIdx.x * 32;
    const int k0 = blockIdx.y * 32;
    const int tx = threadIdx.x & 31;
    const int ty = threadIdx.x >> 5;       // 0..7
#pragma unroll
    for (int i = 0; i < 4; ++i)
        tile[ty*4 + i][tx] = W[(size_t)(k0 + ty*4 + i) * N + n0 + tx];
    __syncthreads();
#pragma unroll
    for (int i = 0; i < 4; ++i)
        Wt[(size_t)(n0 + ty*4 + i) * K + k0 + tx] = (bf16)tile[tx][ty*4 + i];
}

// ------------------------------------------------------------------
// LayerNorm, faithful to reference: Bessel-corrected std, eps on std.
// ------------------------------------------------------------------
__global__ __launch_bounds__(256)
void enc_layernorm_bf16(const float* __restrict__ X, const float* __restrict__ alpha,
                        const float* __restrict__ beta, bf16* __restrict__ H) {
    __shared__ float s1[8], s2[8];
    const int row  = blockIdx.x;
    const int tid  = threadIdx.x;
    const int lane = tid & 31;
    const int w    = tid >> 5;
    const float* x = X + (size_t)row * DM;

    float sum = 0.f, sq = 0.f;
#pragma unroll
    for (int i = 0; i < 4; ++i) {
        float v = x[tid + 256*i];
        sum += v; sq += v*v;
    }
#pragma unroll
    for (int off = 16; off > 0; off >>= 1) {
        sum += __shfl_xor(sum, off, 32);
        sq  += __shfl_xor(sq,  off, 32);
    }
    if (lane == 0) { s1[w] = sum; s2[w] = sq; }
    __syncthreads();
    if (tid == 0) {
        float S = 0.f, Q = 0.f;
#pragma unroll
        for (int i = 0; i < 8; ++i) { S += s1[i]; Q += s2[i]; }
        s1[0] = S; s2[0] = Q;
    }
    __syncthreads();
    const float S  = s1[0], Q = s2[0];
    const float mean = S * (1.0f / DM);
    float var = (Q - (float)DM * mean * mean) * (1.0f / (DM - 1)); // ddof=1
    var = fmaxf(var, 0.f);
    const float inv = 1.0f / (sqrtf(var) + 1e-6f);
    bf16* h = H + (size_t)row * DM;
#pragma unroll
    for (int i = 0; i < 4; ++i) {
        const int c = tid + 256*i;
        h[c] = (bf16)(alpha[c] * (x[c] - mean) * inv + beta[c]);
    }
}

// ------------------------------------------------------------------
// bf16 WMMA GEMM with double-buffered async global->LDS staging.
// C[M,N] = A[M,K](bf16,row-major) x Bt[N,K](bf16, K-contiguous) + bias
// MODE 0: bf16 out, 1: bf16 relu out, 2: f32 out + f32 residual add.
// 128x128 block tile, BK=32, 8 waves (4x2), each wave 32x64.
// ------------------------------------------------------------------
template <int MODE>
__global__ __launch_bounds__(256)
void enc_gemm_bf16(const bf16* __restrict__ A, const bf16* __restrict__ Bt,
                   const float* __restrict__ bias, const float* __restrict__ residual,
                   void* __restrict__ Cout, int M, int N, int K) {
    constexpr int BK = 32, LDT = 40;   // 40-half stride: conflict-free, 16B aligned
    __shared__ bf16 sA[2][128 * LDT];
    __shared__ bf16 sB[2][128 * LDT];

    const int tid  = threadIdx.x;
    const int lane = tid & 31;
    const int w    = tid >> 5;
    const int wm   = w >> 1;            // 0..3 : 32-row slice
    const int wn   = w & 1;             // 0..1 : 64-col slice
    const int m0   = blockIdx.y * 128;
    const int n0   = blockIdx.x * 128;

    // staging geometry: 128 rows x 32 halves = 512 x 16B segs per tile,
    // 2 segs/thread: s = tid and tid+256  ->  rows row0 / row0+64, same seg
    const int row0 = tid >> 2;
    const int seg0 = tid & 3;
    const uint64_t baseA = (uint64_t)A;
    const uint64_t baseB = (uint64_t)Bt;
    uint32_t ldsA[2][2], ldsB[2][2];
#pragma unroll
    for (int b = 0; b < 2; ++b) {
        ldsA[b][0] = lds_offset_of(&sA[b][ row0       * LDT + seg0 * 8]);
        ldsA[b][1] = lds_offset_of(&sA[b][(row0 + 64) * LDT + seg0 * 8]);
        ldsB[b][0] = lds_offset_of(&sB[b][ row0       * LDT + seg0 * 8]);
        ldsB[b][1] = lds_offset_of(&sB[b][(row0 + 64) * LDT + seg0 * 8]);
    }
    auto issue_chunk = [&](int kk, int b) {
        const uint32_t segoff = (uint32_t)(kk + seg0 * 8) * 2u;
        async_copy_b128(ldsA[b][0], (uint32_t)(m0 + row0)      * (uint32_t)K * 2u + segoff, baseA);
        async_copy_b128(ldsA[b][1], (uint32_t)(m0 + row0 + 64) * (uint32_t)K * 2u + segoff, baseA);
        async_copy_b128(ldsB[b][0], (uint32_t)(n0 + row0)      * (uint32_t)K * 2u + segoff, baseB);
        async_copy_b128(ldsB[b][1], (uint32_t)(n0 + row0 + 64) * (uint32_t)K * 2u + segoff, baseB);
    };

    CFrag acc[2][4];
#pragma unroll
    for (int i = 0; i < 2; ++i)
#pragma unroll
        for (int j = 0; j < 4; ++j) acc[i][j].v = zero_v8f();

    issue_chunk(0, 0);
    int cur = 0;
    for (int kk = 0; kk < K; kk += BK, cur ^= 1) {
        wait_async0();          // my chunk-k copies have landed in LDS
        __syncthreads();        // everyone's landed; other buffer free for reuse
        if (kk + BK < K)
            issue_chunk(kk + BK, cur ^ 1);   // overlap next copy with compute

        v16bf af[2], bfr[4];
#pragma unroll
        for (int i = 0; i < 2; ++i)
            af[i] = load_frag16x32(&sA[cur][(wm * 32 + i * 16) * LDT], LDT);
#pragma unroll
        for (int j = 0; j < 4; ++j)
            bfr[j] = load_frag16x32(&sB[cur][(wn * 64 + j * 16) * LDT], LDT);
#pragma unroll
        for (int i = 0; i < 2; ++i)
#pragma unroll
            for (int j = 0; j < 4; ++j)
                acc[i][j].v = __builtin_amdgcn_wmma_f32_16x16x32_bf16(
                    false, af[i], false, bfr[j], (short)0, acc[i][j].v, false, false);
        __syncthreads();        // done reading buf cur before it is refilled
    }

    // epilogue (C layout: lane -> n = L&15, rows m = ((L>=16)?8:0)+r)
    const int nlocal = lane & 15;
    const int mbase  = (lane & 16) ? 8 : 0;
#pragma unroll
    for (int i = 0; i < 2; ++i) {
#pragma unroll
        for (int j = 0; j < 4; ++j) {
            const int gn = n0 + wn * 64 + j * 16 + nlocal;
            const float bv = bias[gn];
#pragma unroll
            for (int r = 0; r < 8; ++r) {
                const int gm = m0 + wm * 32 + i * 16 + mbase + r;
                float v = acc[i][j].e[r] + bv;
                if (MODE == 1) v = v > 0.f ? v : 0.f;
                if (MODE == 2) {
                    ((float*)Cout)[(size_t)gm * N + gn] =
                        v + residual[(size_t)gm * N + gn];
                } else {
                    ((bf16*)Cout)[(size_t)gm * N + gn] = (bf16)v;
                }
            }
        }
    }
}

// ------------------------------------------------------------------
// Attention. Raw-reshape semantics make per-head Q/K/V/O contiguous
// [2048 x 64] blocks at offset (b*16+h)*2048*64 of the flat buffers.
// One workgroup = (head, 16 Q rows). Two-pass softmax through a bf16
// LDS score buffer (normalization folded into the P.V epilogue).
// ------------------------------------------------------------------
#define PSTR 2056            // 2048 + 8 pad (halves)
#define VSTR 264             // 256 + 8 pad (halves)
#define ATTN_LDS_BYTES ((16*PSTR + 64*VSTR) * 2 + (4*32*8 + 16) * 4)

__global__ __launch_bounds__(256)
void enc_attention(const bf16* __restrict__ Q, const bf16* __restrict__ Kb,
                   const bf16* __restrict__ Vb, bf16* __restrict__ O) {
    extern __shared__ char smem[];
    bf16*  P    = (bf16*)smem;                                   // 16 x PSTR
    bf16*  Vt   = (bf16*)(smem + 16 * PSTR * 2);                 // 64 x VSTR
    float* red  = (float*)(smem + (16 * PSTR + 64 * VSTR) * 2);  // 4*32*8
    float* rsum = red + 4 * 32 * 8;                              // 16 recip sums

    const int tid  = threadIdx.x;
    const int lane = tid & 31;
    const int w    = tid >> 5;
    const int q0   = blockIdx.x * 16;                 // 0..2047 step 16
    const size_t hb = (size_t)blockIdx.y * SEQ * DKH; // head base (b*16+h)

    // ---- pass 1: S = (Q K^T) / sqrt(d_model), bf16 into LDS ----
    const v16bf aq0 = load_frag16x32(Q + hb + (size_t)q0 * DKH,      DKH);
    const v16bf aq1 = load_frag16x32(Q + hb + (size_t)q0 * DKH + 32, DKH);
    const float scale = 0.03125f;                     // 1/sqrt(1024)
    const int mbase = (lane & 16) ? 8 : 0;
    for (int j = 0; j < 16; ++j) {                    // this wave's 256 keys
        const int key0 = w * 256 + j * 16;
        const v16bf bk0 = load_frag16x32(Kb + hb + (size_t)key0 * DKH,      DKH);
        const v16bf bk1 = load_frag16x32(Kb + hb + (size_t)key0 * DKH + 32, DKH);
        CFrag c; c.v = zero_v8f();
        c.v = __builtin_amdgcn_wmma_f32_16x16x32_bf16(false, aq0, false, bk0, (short)0, c.v, false, false);
        c.v = __builtin_amdgcn_wmma_f32_16x16x32_bf16(false, aq1, false, bk1, (short)0, c.v, false, false);
#pragma unroll
        for (int r = 0; r < 8; ++r)
            P[(mbase + r) * PSTR + key0 + (lane & 15)] = (bf16)(c.e[r] * scale);
    }
    __syncthreads();

    // ---- pass 2: rowwise max / exp / sum (wave32 shuffles), in place ----
#pragma unroll
    for (int rr = 0; rr < 2; ++rr) {
        const int m = w * 2 + rr;
        bf16* row = P + m * PSTR;
        float mx = -3.0e38f;
        for (int c = lane; c < SEQ; c += 32) mx = fmaxf(mx, (float)row[c]);
#pragma unroll
        for (int off = 16; off > 0; off >>= 1) mx = fmaxf(mx, __shfl_xor(mx, off, 32));
        float sum = 0.f;
        for (int c = lane; c < SEQ; c += 32) {
            const float e = __expf((float)row[c] - mx);
            sum += e;
            row[c] = (bf16)e;                         // unnormalized
        }
#pragma unroll
        for (int off = 16; off > 0; off >>= 1) sum += __shfl_xor(sum, off, 32);
        if (lane == 0) rsum[m] = 1.0f / sum;
    }
    __syncthreads();

    // ---- pass 3: O = P V ; waves: ni = d-slice (4), ki = key half (2) ----
    const int ni = w & 3;
    const int ki = w >> 2;
    CFrag acc; acc.v = zero_v8f();
    for (int kc = 0; kc < 8; ++kc) {                  // 256-key chunks
        __syncthreads();
        {   // stage V chunk transposed: Vt[d][key] (K-contiguous B operand)
            const int key = kc * 256 + tid;
            const bf16* vr = Vb + hb + (size_t)key * DKH;
            if (kc + 1 < 8)
                __builtin_prefetch(Vb + hb + (size_t)(key + 256) * DKH, 0, 1);
#pragma unroll
            for (int d = 0; d < 64; ++d)
                Vt[d * VSTR + tid] = vr[d];
        }
        __syncthreads();
#pragma unroll
        for (int jj = 0; jj < 4; ++jj) {
            const int koff = ki * 128 + jj * 32;
            const v16bf ap = load_frag16x32(P + kc * 256 + koff, PSTR);
            const v16bf bv = load_frag16x32(Vt + (ni * 16) * VSTR + koff, VSTR);
            acc.v = __builtin_amdgcn_wmma_f32_16x16x32_bf16(
                false, ap, false, bv, (short)0, acc.v, false, false);
        }
    }
    if (ki == 1) {
#pragma unroll
        for (int r = 0; r < 8; ++r) red[(ni * 32 + lane) * 8 + r] = acc.e[r];
    }
    __syncthreads();
    if (ki == 0) {
#pragma unroll
        for (int r = 0; r < 8; ++r) {
            const int m = mbase + r;
            const float v = (acc.e[r] + red[(ni * 32 + lane) * 8 + r]) * rsum[m];
            O[hb + (size_t)(q0 + m) * DKH + ni * 16 + (lane & 15)] = (bf16)v;
        }
    }
}

// ------------------------------------------------------------------
// Host-side orchestration. Workspace layout (80 MB total):
//   [0,8)MB   hb    (LN output, reused for LN2 output)
//   [8,16)MB  Wq/Wk/Wv/Wo transposed bf16 (2MB each)
//   [16,24)   W1t   [24,32) W2t
//   [32,40)q [40,48)k [48,56)v [56,64)o  -> [32,64) reused as ff1
//   [64,80)   x1 fp32 residual
// ------------------------------------------------------------------
extern "C" void kernel_launch(void* const* d_in, const int* in_sizes, int n_in,
                              void* d_out, int out_size, void* d_ws, size_t ws_size,
                              hipStream_t stream) {
    (void)in_sizes; (void)n_in; (void)out_size; (void)ws_size;
    const float* x    = (const float*)d_in[0];
    const float* Wq   = (const float*)d_in[1];
    const float* bq   = (const float*)d_in[2];
    const float* Wk   = (const float*)d_in[3];
    const float* bk   = (const float*)d_in[4];
    const float* Wv   = (const float*)d_in[5];
    const float* bv   = (const float*)d_in[6];
    const float* Wo   = (const float*)d_in[7];
    const float* bo   = (const float*)d_in[8];
    const float* W1   = (const float*)d_in[9];
    const float* b1   = (const float*)d_in[10];
    const float* W2   = (const float*)d_in[11];
    const float* b2   = (const float*)d_in[12];
    const float* ln1a = (const float*)d_in[13];
    const float* ln1b = (const float*)d_in[14];
    const float* ln2a = (const float*)d_in[15];
    const float* ln2b = (const float*)d_in[16];

    char* ws = (char*)d_ws;
    const size_t MB = (size_t)1 << 20;
    bf16*  hb  = (bf16*)(ws + 0);
    bf16*  Wqt = (bf16*)(ws + 8  * MB);
    bf16*  Wkt = (bf16*)(ws + 10 * MB);
    bf16*  Wvt = (bf16*)(ws + 12 * MB);
    bf16*  Wot = (bf16*)(ws + 14 * MB);
    bf16*  W1t = (bf16*)(ws + 16 * MB);
    bf16*  W2t = (bf16*)(ws + 24 * MB);
    bf16*  qb  = (bf16*)(ws + 32 * MB);
    bf16*  kb  = (bf16*)(ws + 40 * MB);
    bf16*  vb  = (bf16*)(ws + 48 * MB);
    bf16*  ob  = (bf16*)(ws + 56 * MB);
    bf16*  ff1 = (bf16*)(ws + 32 * MB);   // reuse q/k/v/o region after Wo GEMM
    float* x1  = (float*)(ws + 64 * MB);

    // weight conversion + transpose (bf16, K-contiguous)
    enc_transpose_bf16<<<dim3(DM/32,  DM/32),  256, 0, stream>>>(Wq, Wqt, DM,  DM);
    enc_transpose_bf16<<<dim3(DM/32,  DM/32),  256, 0, stream>>>(Wk, Wkt, DM,  DM);
    enc_transpose_bf16<<<dim3(DM/32,  DM/32),  256, 0, stream>>>(Wv, Wvt, DM,  DM);
    enc_transpose_bf16<<<dim3(DM/32,  DM/32),  256, 0, stream>>>(Wo, Wot, DM,  DM);
    enc_transpose_bf16<<<dim3(DFF/32, DM/32),  256, 0, stream>>>(W1, W1t, DM,  DFF);
    enc_transpose_bf16<<<dim3(DM/32,  DFF/32), 256, 0, stream>>>(W2, W2t, DFF, DM);

    // sublayer 1: pre-norm MHA + residual
    enc_layernorm_bf16<<<ROWS, 256, 0, stream>>>(x, ln1a, ln1b, hb);
    const dim3 gDD(DM / 128, ROWS / 128);             // (8, 32)
    enc_gemm_bf16<0><<<gDD, 256, 0, stream>>>(hb, Wqt, bq, nullptr, qb, ROWS, DM, DM);
    enc_gemm_bf16<0><<<gDD, 256, 0, stream>>>(hb, Wkt, bk, nullptr, kb, ROWS, DM, DM);
    enc_gemm_bf16<0><<<gDD, 256, 0, stream>>>(hb, Wvt, bv, nullptr, vb, ROWS, DM, DM);
    enc_attention<<<dim3(SEQ / 16, BATCH * NH), 256, ATTN_LDS_BYTES, stream>>>(qb, kb, vb, ob);
    enc_gemm_bf16<2><<<gDD, 256, 0, stream>>>(ob, Wot, bo, x, x1, ROWS, DM, DM);

    // sublayer 2: pre-norm FFN + residual
    enc_layernorm_bf16<<<ROWS, 256, 0, stream>>>(x1, ln2a, ln2b, hb);
    enc_gemm_bf16<1><<<dim3(DFF / 128, ROWS / 128), 256, 0, stream>>>(
        hb, W1t, b1, nullptr, ff1, ROWS, DFF, DM);
    enc_gemm_bf16<2><<<gDD, 256, 0, stream>>>(ff1, W2t, b2, x1, (float*)d_out, ROWS, DM, DFF);
}
